// QuantumVelocityField_26972394619637
// MI455X (gfx1250) — compile-verified, matrix operand order checked by reading, not run on hardware
//
#include <hip/hip_runtime.h>
#include <hip/hip_bf16.h>

typedef __attribute__((ext_vector_type(16))) _Float16 v16h;
typedef __attribute__((ext_vector_type(8)))  float    v8f;
typedef __attribute__((ext_vector_type(2)))  float    v2f;

#define WAVES_PER_BLOCK 4
#define ROWS_PER_WAVE   16
#define ROWS_PER_BLOCK  (WAVES_PER_BLOCK * ROWS_PER_WAVE)   // 64 rows per tile-iter
#define ITERS           4                                    // tiles per block

__device__ __forceinline__ float silu_f(float x) {
  // x * sigmoid(x): v_exp_f32 + v_rcp_f32, no IEEE division sequence
  return x * __builtin_amdgcn_rcpf(1.0f + __expf(-x));
}

__global__ __launch_bounds__(128)
void QuantumVelocityField_kernel(const float* __restrict__ t,
                                 const float* __restrict__ bloch,
                                 const float* __restrict__ W1,
                                 const float* __restrict__ b1,
                                 const float* __restrict__ W2,
                                 const float* __restrict__ b2,
                                 const float* __restrict__ W3,
                                 const float* __restrict__ b3,
                                 float* __restrict__ out)
{
  // ---- LDS (~57 KB) ----
  __shared__ float    w1s[4 * 128];
  __shared__ float    b1s[128];
  __shared__ float    b2s[128];
  __shared__ float    b3s[8];
  __shared__ _Float16 w2s[4 * 8 * 32 * 16];                    // W2 B-frags (32 KB)
  __shared__ _Float16 w3b[4 * 32 * 16];                        // W3 B-frags, cols 6..15 = 0
  __shared__ _Float16 hba[WAVES_PER_BLOCK * 4 * 32 * 16];      // h1/h2 A-frag buffer (16 KB, reused)
  __shared__ float    prm[WAVES_PER_BLOCK][ROWS_PER_WAVE][8];  // layer-3 params

  const int tid  = threadIdx.x;
  const int wave = tid >> 5;
  const int lane = tid & 31;
  const int g    = lane >> 4;   // half-wave group
  const int nl   = lane & 15;

  // ---- stage weights (once per block, amortized over ITERS*64 rows) ----
  for (int i = tid; i < 4 * 128; i += 128) w1s[i] = W1[i];
  if (tid < 128) { b1s[tid] = b1[tid]; b2s[tid] = b2[tid]; }
  if (tid < 8)   b3s[tid] = (tid < 6) ? b3[tid] : 0.0f;
  // W2 [K=128][N=128]: coalesced source reads, forward-swizzled B-frag stores.
  // B-frag slot of (K,N): kt=K>>5, kin=K&31, gg=kin bit3, h=(kin&7)+((kin>>4)<<3),
  //                       nt=N>>4, lane=gg*16+(N&15)
  for (int i = tid; i < 128 * 128; i += 128) {
    const int K   = i >> 7, n = i & 127;
    const int kin = K & 31;
    const int h   = (kin & 7) + ((kin >> 4) << 3);
    const int ln  = ((kin >> 3) & 1) * 16 + (n & 15);
    const int nt  = n >> 4, kt = K >> 5;
    w2s[(((kt << 3) + nt) * 32 + ln) * 16 + h] = (_Float16)W2[i];
  }
  // W3 [K=128][N=6] -> B-frags padded to N=16 with zeros
  for (int i = tid; i < 4 * 32 * 16; i += 128) {
    const int h  = i & 15;
    const int ln = (i >> 4) & 31;
    const int kt = i >> 9;
    const int gg = ln >> 4;
    const int n  = ln & 15;
    const int kk = (h & 7) + ((h >> 3) << 4) + gg * 8;
    w3b[i] = (n < 6) ? (_Float16)W3[(kt * 32 + kk) * 6 + n] : (_Float16)0.0f;
  }
  __syncthreads();   // only barrier in the kernel

  // W1 B-fragments for wmma_f32_16x16x4_f32, kept in registers:
  // 32-bit B 4x16 layout: lane (gg,nl) holds col N=nl, {K=gg*2+0, K=gg*2+1}
  v2f w1bf[8];
#pragma unroll
  for (int nt = 0; nt < 8; ++nt) {
    const int c = nt * 16 + nl;
    w1bf[nt][0] = w1s[(g * 2 + 0) * 128 + c];
    w1bf[nt][1] = w1s[(g * 2 + 1) * 128 + c];
  }

  const int r      = nl;                 // tile row (both half-waves cover rows 0..15)
  const int hwbase = wave * 4 * 32 * 16; // wave-private A-frag buffer
  const int ga     = (nl >> 3) & 1;      // target A-group for D->A scatter

#pragma unroll 1
  for (int it = 0; it < ITERS; ++it) {
    const int rowg = ((blockIdx.x * ITERS + it) * WAVES_PER_BLOCK + wave) * ROWS_PER_WAVE + r;

    // ---- layer 1: [16x4] @ [4x16] via v_wmma_f32_16x16x4_f32 (fp32 exact) ----
    // 32-bit A 16x4 layout: lanes 0-15 hold {K0,K1} of row, lanes 16-31 {K2,K3}
    v2f af1;
    if (g == 0) { af1[0] = t[rowg];             af1[1] = bloch[rowg * 3 + 0]; }
    else        { af1[0] = bloch[rowg * 3 + 1]; af1[1] = bloch[rowg * 3 + 2]; }

#pragma unroll
    for (int nt = 0; nt < 8; ++nt) {
      v8f z = {};
      z = __builtin_amdgcn_wmma_f32_16x16x4_f32(
          false, af1, false, w1bf[nt], (short)0, z, false, false);
      // bias + SiLU on D-frag, scatter into A-frag-swizzled LDS:
      //   D elem: row m=g*8+v, col c=nt*16+nl -> kt=nt>>1, h=(nl&7)+((nt&1)<<3), lane=ga*16+m
      const float bb = b1s[nt * 16 + nl];
      const int   kt = nt >> 1;
      const int   hh = (nl & 7) + ((nt & 1) << 3);
#pragma unroll
      for (int v = 0; v < 8; ++v) {
        hba[hwbase + (kt * 32 + ga * 16 + g * 8 + v) * 16 + hh] =
            (_Float16)silu_f(z[v] + bb);
      }
    }

    // ---- layer 2: [16x128] @ [128x128], 32x v_wmma_f32_16x16x32_f16 ----
    v16h afrag[4];
#pragma unroll
    for (int kt = 0; kt < 4; ++kt)
      afrag[kt] = *(const v16h*)(&hba[hwbase + (kt * 32 + lane) * 16]);

    v8f acc[8] = {};
#pragma unroll
    for (int kt = 0; kt < 4; ++kt) {
#pragma unroll
      for (int nt = 0; nt < 8; ++nt) {
        const v16h bfrag = *(const v16h*)(&w2s[(((kt << 3) + nt) * 32 + lane) * 16]);
        acc[nt] = __builtin_amdgcn_wmma_f32_16x16x32_f16(
            false, afrag[kt], false, bfrag, (short)0, acc[nt], false, false);
      }
    }

    // ---- bias + SiLU, scatter h2 into the same A-frag buffer (wave-private) ----
#pragma unroll
    for (int nt = 0; nt < 8; ++nt) {
      const float bb = b2s[nt * 16 + nl];
      const int   kt = nt >> 1;
      const int   hh = (nl & 7) + ((nt & 1) << 3);
#pragma unroll
      for (int v = 0; v < 8; ++v) {
        hba[hwbase + (kt * 32 + ga * 16 + g * 8 + v) * 16 + hh] =
            (_Float16)silu_f(acc[nt][v] + bb);
      }
    }

    // ---- layer 3: [16x128] @ [128x16(pad of 6)], 4 WMMAs ----
    v8f pacc = {};
#pragma unroll
    for (int kt = 0; kt < 4; ++kt) {
      const v16h a3  = *(const v16h*)(&hba[hwbase + (kt * 32 + lane) * 16]);
      const v16h b3f = *(const v16h*)(&w3b[(kt * 32 + lane) * 16]);
      pacc = __builtin_amdgcn_wmma_f32_16x16x32_f16(
          false, a3, false, b3f, (short)0, pacc, false, false);
    }

    // D: lane (g,nl) vgpr v holds param (row m=g*8+v, j=nl); keep j<6
    if (nl < 6) {
      const float bj = b3s[nl];
#pragma unroll
      for (int v = 0; v < 8; ++v) prm[wave][g * 8 + v][nl] = pacc[v] + bj;
    }

    // ---- quantum circuit: lane r of low half-wave owns row r ----
    // (same-wave LDS store->load is in-order; no barrier needed)
    if (g == 0) {
      const float th[3] = {prm[wave][r][0], prm[wave][r][2], prm[wave][r][4]};
      const float ph[3] = {prm[wave][r][1], prm[wave][r][3], prm[wave][r][5]};
      float ar = 1.0f, ai = 0.0f, br = 0.0f, bi = 0.0f;
#pragma unroll
      for (int l = 0; l < 3; ++l) {
        const float c = __cosf(0.5f * th[l]);
        const float s = __sinf(0.5f * th[l]);
        const float a1r = c * ar - s * br, a1i = c * ai - s * bi;
        const float b1r = s * ar + c * br, b1i = s * ai + c * bi;
        const float ec = __cosf(0.5f * ph[l]);
        const float es = -__sinf(0.5f * ph[l]);              // e = ec + i*es
        ar = ec * a1r - es * a1i;  ai = ec * a1i + es * a1r; // e * a1
        br = ec * b1r + es * b1i;  bi = ec * b1i - es * b1r; // conj(e) * b1
      }
      const float zr = ar * br + ai * bi;   // conj(a)*b
      const float zi = ar * bi - ai * br;
      out[rowg * 3 + 0] = 2.0f * zr;
      out[rowg * 3 + 1] = 2.0f * zi;
      out[rowg * 3 + 2] = (ar * ar + ai * ai) - (br * br + bi * bi);
    }
  }
}

extern "C" void kernel_launch(void* const* d_in, const int* in_sizes, int n_in,
                              void* d_out, int out_size, void* d_ws, size_t ws_size,
                              hipStream_t stream) {
  const float* t     = (const float*)d_in[0];
  const float* bloch = (const float*)d_in[1];
  const float* W1    = (const float*)d_in[2];
  const float* b1    = (const float*)d_in[3];
  const float* W2    = (const float*)d_in[4];
  const float* b2    = (const float*)d_in[5];
  const float* W3    = (const float*)d_in[6];
  const float* b3    = (const float*)d_in[7];
  float* out = (float*)d_out;

  const int B = in_sizes[0];                             // 262144
  const int grid = B / (ROWS_PER_BLOCK * ITERS);         // 1024 blocks of 128 threads
  QuantumVelocityField_kernel<<<grid, 128, 0, stream>>>(
      t, bloch, W1, b1, W2, b2, W3, b3, out);
}
